// CrossAttention_86878598463956
// MI455X (gfx1250) — matrix operand order repaired and checked by reading, not compiled, not run on hardware
//
#include <hip/hip_runtime.h>
#include <hip/hip_bf16.h>

typedef __bf16 bf16_t;
typedef __bf16 v16bf __attribute__((ext_vector_type(16)));
typedef __bf16 v8bf  __attribute__((ext_vector_type(8)));
typedef __bf16 v2bf  __attribute__((ext_vector_type(2)));
typedef float  v8f   __attribute__((ext_vector_type(8)));

#define WMMA_BF16(A_, B_, C_) \
  __builtin_amdgcn_wmma_f32_16x16x32_bf16(false, (A_), false, (B_), (short)0, (C_), false, false)

// ---------------------------------------------------------------------------
// Vectorized WMMA fragment loaders (gfx1250 wave32 layouts, CDNA5 ISA §7.12.2)
// All LDS layouts are arranged so each lane's fragment data is contiguous and
// 16B-aligned -> ds_load_b128, no scalar u16 traffic.
// ---------------------------------------------------------------------------

// A fragment 16x32: logical A[m,k] = s[(row0+m)*ld + k]; ld % 8 == 0, k0 % 8 == 0.
// Lanes 0-15: M=lane, K {0..7,16..23}; lanes 16-31: same M, K {8..15,24..31}.
__device__ inline v16bf load_a_frag(const bf16_t* s, int row0, int k0, int ld) {
  const int lane = threadIdx.x & 31;
  const int hlf  = lane >> 4;
  const bf16_t* p = s + (row0 + (lane & 15)) * ld + k0 + hlf * 8;
  union { v16bf v; v8bf h[2]; } f;
  f.h[0] = *(const v8bf*)p;
  f.h[1] = *(const v8bf*)(p + 16);
  return f.v;
}

// B fragment 32x16 from N-major storage: logical B[k,n] = s[(n0+n)*ld + k].
// Lanes 0-15: N=lane, K {0..7,16..23}; lanes 16-31: K {8..15,24..31}.
__device__ inline v16bf load_b_frag(const bf16_t* s, int k0, int n0, int ld) {
  const int lane = threadIdx.x & 31;
  const int hlf  = lane >> 4;
  const bf16_t* p = s + (n0 + (lane & 15)) * ld + k0 + hlf * 8;
  union { v16bf v; v8bf h[2]; } f;
  f.h[0] = *(const v8bf*)p;
  f.h[1] = *(const v8bf*)(p + 16);
  return f.v;
}

// ---------------------------------------------------------------------------
// Generic bf16-WMMA GEMM: C[b] = alpha * A(MxK,f32) * B[b](KxN) (+ bias)
// BM=64 BN=128 BK=32, 256 threads = 8 waves, 2x2 16x16 tiles per wave.
// B tile staged N-major in LDS so B fragments are contiguous loads.
// TRANSC: store C transposed (C[n][m], ld=M) with one 16B store per tile/lane.
// ---------------------------------------------------------------------------
template <typename TB, bool TRANSB, bool TRANSC, bool OUT_BF16>
__global__ __launch_bounds__(256) void gemm_wmma_kernel(
    const float* __restrict__ A, const TB* __restrict__ B, void* __restrict__ Cout,
    const float* __restrict__ bias, float alpha, int M, int N, int K,
    long strideB, long strideC) {
  constexpr int BM = 64, BN = 128, BK = 32, BKP = BK + 8;
  __shared__ bf16_t As[BM][BKP];    // row-major (K contiguous)
  __shared__ bf16_t Bst[BN][BKP];   // N-major   (K contiguous)

  const int tid  = threadIdx.x;
  const int wave = tid >> 5;
  const int lane = tid & 31;
  const int hlf  = lane >> 4;
  const int m0 = blockIdx.y * BM;
  const int n0 = blockIdx.x * BN;
  const int bb = blockIdx.z;
  const TB* Bb = B + (long)bb * strideB;

  const int wm = (wave & 1) * 32;   // 2 M tiles: wm, wm+16
  const int wn = (wave >> 1) * 32;  // 2 N tiles: wn, wn+16

  v8f acc[2][2];
#pragma unroll
  for (int mt = 0; mt < 2; ++mt)
#pragma unroll
    for (int nt = 0; nt < 2; ++nt)
#pragma unroll
      for (int r = 0; r < 8; ++r) acc[mt][nt][r] = 0.0f;

  for (int k0 = 0; k0 < K; k0 += BK) {
    // A tile 64x32 f32 -> bf16, paired stores
#pragma unroll
    for (int i = 0; i < (BM * BK) / 512; ++i) {  // 4 pairs/thread
      int pe = tid + i * 256;
      int r = pe >> 4, c2 = (pe & 15) * 2;
      v2bf w;
      w[0] = (bf16_t)A[(long)(m0 + r) * K + k0 + c2];
      w[1] = (bf16_t)A[(long)(m0 + r) * K + k0 + c2 + 1];
      *(v2bf*)&As[r][c2] = w;
    }
    // B tile 32x128 -> N-major LDS
    if (TRANSB) {  // source Bsrc[n][k], k contiguous: paired copy
#pragma unroll
      for (int i = 0; i < (BK * BN) / 512; ++i) {  // 8 pairs/thread
        int pe = tid + i * 256;
        int n = pe >> 4, k2 = (pe & 15) * 2;
        v2bf w;
        w[0] = (bf16_t)(float)Bb[(long)(n0 + n) * K + k0 + k2];
        w[1] = (bf16_t)(float)Bb[(long)(n0 + n) * K + k0 + k2 + 1];
        *(v2bf*)&Bst[n][k2] = w;
      }
    } else {       // source B[k][n], n contiguous reads, transposed scalar writes
#pragma unroll
      for (int i = 0; i < (BK * BN) / 256; ++i) {  // 16/thread
        int e = tid + i * 256;
        int k = e >> 7, n = e & 127;
        Bst[n][k] = (bf16_t)(float)Bb[(long)(k0 + k) * N + (n0 + n)];
      }
    }
    __syncthreads();

    v16bf a0 = load_a_frag(&As[0][0], wm, 0, BKP);
    v16bf a1 = load_a_frag(&As[0][0], wm + 16, 0, BKP);
    v16bf b0 = load_b_frag(&Bst[0][0], 0, wn, BKP);
    v16bf b1 = load_b_frag(&Bst[0][0], 0, wn + 16, BKP);
    acc[0][0] = WMMA_BF16(a0, b0, acc[0][0]);
    acc[0][1] = WMMA_BF16(a0, b1, acc[0][1]);
    acc[1][0] = WMMA_BF16(a1, b0, acc[1][0]);
    acc[1][1] = WMMA_BF16(a1, b1, acc[1][1]);
    __syncthreads();
  }

#pragma unroll
  for (int mt = 0; mt < 2; ++mt)
#pragma unroll
    for (int nt = 0; nt < 2; ++nt) {
      int nn = n0 + wn + nt * 16 + (lane & 15);
      int mbase = m0 + wm + mt * 16 + hlf * 8;
      if (TRANSC) {  // C[n][m]: 8 consecutive m per lane -> one 16B store
        v8bf o8;
#pragma unroll
        for (int r = 0; r < 8; ++r)
          o8[r] = (bf16_t)(acc[mt][nt][r] * alpha + (bias ? bias[mbase + r] : 0.0f));
        *(v8bf*)((bf16_t*)Cout + (long)bb * strideC + (long)nn * M + mbase) = o8;
      } else {
#pragma unroll
        for (int r = 0; r < 8; ++r) {
          float v = acc[mt][nt][r] * alpha + (bias ? bias[mbase + r] : 0.0f);
          if (OUT_BF16)
            ((bf16_t*)Cout)[(long)bb * strideC + (long)(mbase + r) * N + nn] = (bf16_t)v;
          else
            ((float*)Cout)[(long)bb * strideC + (long)(mbase + r) * N + nn] = v;
        }
      }
    }
}

// ---------------------------------------------------------------------------
// Attention: one workgroup = (batch b, head h, 128 query rows), 8 waves.
// LDS (157 KB, CDNA5 WGP has 320 KB):
//   sQi [128][72]  query-major Q  -> A frags contiguous
//   sKt [256][72]  t-major K      -> B frags contiguous (S = Q K^T)
//   sV  [ 64][264] d-major V      -> B frags contiguous (O = P V^T)
//   sP  8x[16][264] per-wave P    -> A frags contiguous
// ---------------------------------------------------------------------------
__global__ __launch_bounds__(256) void attn_kernel(
    const bf16_t* __restrict__ qbuf,   // (B, 4096, 512) bf16, pre-scaled
    const bf16_t* __restrict__ kvbuf,  // (B, 1024, 256) bf16: K rows then V rows
    bf16_t* __restrict__ obuf) {       // (B, 512, 4096) bf16 (h*64+d, i)
  constexpr int T = 256, NQ = 4096, HID = 512;
  constexpr int LDD = 72;    // 64 + 8
  constexpr int LDT = 264;   // 256 + 8
  extern __shared__ char smem_raw[];
  bf16_t* sQi = (bf16_t*)smem_raw;       // 128*72
  bf16_t* sKt = sQi + 128 * LDD;         // 256*72
  bf16_t* sV  = sKt + 256 * LDD;         // 64*264
  bf16_t* sP  = sV + 64 * LDT;           // 8*16*264

  const int tid  = threadIdx.x;
  const int wave = tid >> 5;
  const int lane = tid & 31;
  const int hlf  = lane >> 4;
  const int i0 = blockIdx.x * 128;
  const int h  = blockIdx.y;
  const int b  = blockIdx.z;

  const bf16_t* kptr = kvbuf + ((long)b * 1024 + h * 64) * T;
  const bf16_t* vptr = kptr + (long)HID * T;
  const bf16_t* qptr = qbuf + ((long)b * NQ + i0) * HID + h * 64;

  // Q block 128x64: contiguous 16B copies
#pragma unroll
  for (int i = 0; i < 4; ++i) {
    int e = tid + i * 256;          // 1024 v8 chunks
    int row = e >> 3, v = (e & 7) * 8;
    *(v8bf*)&sQi[row * LDD + v] = *(const v8bf*)(qptr + (long)row * HID + v);
  }
  // K 64x256 -> t-major (transposed scalar writes, coalesced reads)
  for (int i = 0; i < 64; ++i)
    sKt[tid * LDD + i] = kptr[(long)i * T + tid];
  // V 64x256: contiguous 16B copies
#pragma unroll
  for (int i = 0; i < 8; ++i) {
    int e = tid + i * 256;          // 2048 v8 chunks
    int d = e >> 5, v = (e & 31) * 8;
    *(v8bf*)&sV[d * LDT + v] = *(const v8bf*)(vptr + (long)d * T + v);
  }
  __syncthreads();

  // S = Q K^T for this wave's 16 rows (K-dim 64 -> 2 WMMA steps, 16 j tiles)
  const int mq = wave * 16;
  v16bf aq0 = load_a_frag(sQi, mq, 0, LDD);
  v16bf aq1 = load_a_frag(sQi, mq, 32, LDD);

  v8f s[16];
#pragma unroll
  for (int jt = 0; jt < 16; ++jt) {
    v8f a;
#pragma unroll
    for (int r = 0; r < 8; ++r) a[r] = 0.0f;
    a = WMMA_BF16(aq0, load_b_frag(sKt, 0, jt * 16, LDD), a);
    a = WMMA_BF16(aq1, load_b_frag(sKt, 32, jt * 16, LDD), a);
    s[jt] = a;
  }

  // Row softmax over 256 keys; row m = hlf*8+r lives in 16 lanes of one half.
  float rinv[8];
  bf16_t* sPw = sP + wave * 16 * LDT;
#pragma unroll
  for (int r = 0; r < 8; ++r) {
    float mx = -3.0e38f;
#pragma unroll
    for (int jt = 0; jt < 16; ++jt) mx = fmaxf(mx, s[jt][r]);
#pragma unroll
    for (int off = 1; off < 16; off <<= 1) mx = fmaxf(mx, __shfl_xor(mx, off, 32));
    float sum = 0.0f;
#pragma unroll
    for (int jt = 0; jt < 16; ++jt) {
      float p = __expf(s[jt][r] - mx);
      sum += p;
      sPw[(hlf * 8 + r) * LDT + jt * 16 + (lane & 15)] = (bf16_t)p;  // D->A relayout
    }
#pragma unroll
    for (int off = 1; off < 16; off <<= 1) sum += __shfl_xor(sum, off, 32);
    rinv[r] = 1.0f / sum;
  }

  // P slab is wave-private: CDNA5 split-counter wait on our own DS stores.
  asm volatile("s_wait_dscnt 0" ::: "memory");

  // O = P V^T (K-dim 256 -> 8 steps, 4 d tiles)
  v8f o[4];
#pragma unroll
  for (int t = 0; t < 4; ++t)
#pragma unroll
    for (int r = 0; r < 8; ++r) o[t][r] = 0.0f;
#pragma unroll
  for (int kt = 0; kt < 8; ++kt) {
    v16bf ap = load_a_frag(sPw, 0, kt * 32, LDT);
#pragma unroll
    for (int dt = 0; dt < 4; ++dt)
      o[dt] = WMMA_BF16(ap, load_b_frag(sV, kt * 32, dt * 16, LDT), o[dt]);
  }

  // Store normalized O as (h*64+d, i) bf16 for the output projection.
  bf16_t* optr = obuf + ((long)b * HID + h * 64) * NQ + i0 + mq;
#pragma unroll
  for (int dt = 0; dt < 4; ++dt) {
    int d = dt * 16 + (lane & 15);
#pragma unroll
    for (int r = 0; r < 8; ++r) {
      int m = hlf * 8 + r;
      optr[(long)d * NQ + m] = (bf16_t)(o[dt][r] * rinv[r]);
    }
  }
}

// ---------------------------------------------------------------------------
extern "C" void kernel_launch(void* const* d_in, const int* in_sizes, int n_in,
                              void* d_out, int out_size, void* d_ws, size_t ws_size,
                              hipStream_t stream) {
  (void)in_sizes; (void)n_in; (void)out_size; (void)ws_size;
  const float* x    = (const float*)d_in[0];  // (8,512,64,64)
  const float* ctx  = (const float*)d_in[1];  // (8,256,768)
  const float* Wq   = (const float*)d_in[2];  // (512,512)
  const float* Wkv  = (const float*)d_in[3];  // (1024,768)
  const float* Wout = (const float*)d_in[4];  // (512,512)
  const float* bout = (const float*)d_in[5];  // (512,)
  float* y = (float*)d_out;

  const int B = 8, C = 512, NQ = 4096, T = 256, CD = 768, HID = 512;

  char* ws = (char*)d_ws;
  bf16_t* qbuf  = (bf16_t*)ws; ws += (size_t)B * NQ * HID * 2;      // (B,4096,512)
  bf16_t* kvbuf = (bf16_t*)ws; ws += (size_t)B * 2 * HID * T * 2;   // (B,1024,256)
  bf16_t* obuf  = (bf16_t*)ws;                                      // (B,512,4096)

  // 1) KV projection: kv[b][o][t] = sum_c Wkv[o][c] * ctx[b][t][c]
  gemm_wmma_kernel<float, true, false, true>
      <<<dim3(T / 128, 1024 / 64, B), 256, 0, stream>>>(
          Wkv, ctx, kvbuf, nullptr, 1.0f, 1024, T, CD, (long)T * CD, (long)1024 * T);

  // 2) Q projection, pre-scaled, stored transposed (B, NQ, HID)
  gemm_wmma_kernel<float, false, true, true>
      <<<dim3(NQ / 128, HID / 64, B), 256, 0, stream>>>(
          Wq, x, qbuf, nullptr, 0.125f, HID, NQ, C, (long)C * NQ, (long)NQ * HID);

  // 3) Attention (157 KB dynamic LDS; CDNA5 WGP has 320 KB)
  int smem = (int)((128 * 72 + 256 * 72 + 64 * 264 + 8 * 16 * 264) * sizeof(bf16_t));
  hipFuncSetAttribute((const void*)attn_kernel,
                      hipFuncAttributeMaxDynamicSharedMemorySize, smem);
  attn_kernel<<<dim3(NQ / 128, 8, B), 256, smem, stream>>>(qbuf, kvbuf, obuf);

  // 4) Output projection + bias -> f32 result
  gemm_wmma_kernel<bf16_t, false, false, false>
      <<<dim3(NQ / 128, C / 64, B), 256, 0, stream>>>(
          Wout, obuf, y, bout, 1.0f, C, NQ, HID, (long)HID * NQ, (long)C * NQ);
}
